// GPSConv_24799141167758
// MI455X (gfx1250) — matrix operand order repaired and checked by reading, not compiled, not run on hardware
//
#include <hip/hip_runtime.h>
#include <hip/hip_bf16.h>

#define NTOT 32768
#define CH   128
#define BGR  64
#define NPG  512
#define NH   4
#define DHD  32
#define NEDGE 524288
#define EPSBN 1e-5f

typedef __attribute__((ext_vector_type(16))) _Float16 v16h;
typedef __attribute__((ext_vector_type(8)))  float    v8f;
typedef __attribute__((__vector_size__(4 * sizeof(int)))) int v4i;

union V16H { v16h v; uint4 q[2]; };

#if __has_builtin(__builtin_amdgcn_global_load_async_to_lds_b128)
#define ATTN_ASYNC_V 1
// builtin signature (from clang diagnostics): (v4i __device__*, v4i __shared__*, Imm, Imm)
#define GPTR(p) ((__attribute__((address_space(1))) v4i*)(const_cast<_Float16*>(p)))
#define LPTR(p) ((__attribute__((address_space(3))) v4i*)(p))
#endif

// ---- WMMA operand loaders (CDNA5 16-bit layouts, wave32) ----
// A operand: 16x32 f16 tile, memory layout [row][k] with stride ld (k contiguous)
__device__ __forceinline__ v16h load_A16(const _Float16* base, int ld) {
    int lane = threadIdx.x & 31;
    const _Float16* p = base + (size_t)(lane & 15) * ld + ((lane >> 4) << 3);
    V16H r;
    r.q[0] = *(const uint4*)(p);        // K = koff .. koff+7
    r.q[1] = *(const uint4*)(p + 16);   // K = 16+koff .. 16+koff+7
    return r.v;
}
// B operand: 32x16 f16 tile, memory layout [n][k] with stride ld (k contiguous)
__device__ __forceinline__ v16h load_B16(const _Float16* base, int ld) {
    int lane = threadIdx.x & 31;
    const _Float16* p = base + (size_t)(lane & 15) * ld + ((lane >> 4) << 3);
    V16H r;
    r.q[0] = *(const uint4*)(p);        // K = koff .. koff+7
    r.q[1] = *(const uint4*)(p + 16);   // K = 16+koff .. 16+koff+7
    return r.v;
}

__device__ __forceinline__ v8f wmma_f16(v16h a, v16h b, v8f c) {
    return __builtin_amdgcn_wmma_f32_16x16x32_f16(false, a, false, b, (short)0, c, false, false);
}

// ---- elementwise f32 -> f16 convert ----
__global__ void k_cvt_f16(const float* __restrict__ in, _Float16* __restrict__ out, int n) {
    int i = blockIdx.x * blockDim.x + threadIdx.x;
    if (i < n) out[i] = (_Float16)in[i];
}

// ---- edge scatter: agg[dst] += x[src], cnt[dst] += 1 ----
__global__ void k_scatter(const int* __restrict__ ei, const float* __restrict__ x,
                          float* __restrict__ agg, float* __restrict__ cnt) {
    int tid = blockIdx.x * blockDim.x + threadIdx.x;     // E*4 threads
    if (tid >= NEDGE * 4) return;
    int e = tid >> 2, part = tid & 3;
    int src = ei[e], dst = ei[NEDGE + e];
    const float* xr = x + (size_t)src * CH + part * 32;
    float* ar = agg + (size_t)dst * CH + part * 32;
    #pragma unroll
    for (int j = 0; j < 32; ++j) atomicAdd(ar + j, xr[j]);
    if (part == 0) atomicAdd(cnt + dst, 1.0f);
}

// ---- mean + convert to f16 ----
__global__ void k_agg_fin(const float* __restrict__ agg, const float* __restrict__ cnt,
                          _Float16* __restrict__ ah) {
    int i = blockIdx.x * blockDim.x + threadIdx.x;       // NT*128
    if (i >= NTOT * CH) return;
    float c = fmaxf(cnt[i >> 7], 1.0f);
    ah[i] = (_Float16)(agg[i] / c);
}

// ---- generic WMMA GEMM: out[M,N] = A[M,K] * W[N,K]^T + bias (+resid) ----
// KT = K/32 k-tiles. One wave computes a 16-row tile across all N.
template<int KT>
__global__ void k_gemm(const _Float16* __restrict__ A, int lda,
                       const _Float16* __restrict__ W, int ldw,
                       const float* __restrict__ bias,
                       const float* __restrict__ resid,   // nullable, stride == N (N==128 cases)
                       float* __restrict__ outF32,        // nullable
                       _Float16* __restrict__ outF16,     // nullable
                       _Float16* __restrict__ vt,         // nullable: qkv mode, V -> [b,h,dh,n]
                       int N, int relu) {
    int wave = threadIdx.x >> 5;
    int lane = threadIdx.x & 31;
    int mtile = blockIdx.x * (blockDim.x >> 5) + wave;
    int mbase = mtile * 16;
    if (mbase >= NTOT) return;

    v16h a[KT];
    const _Float16* ab = A + (size_t)mbase * lda;
    #pragma unroll
    for (int k = 0; k < KT; ++k) a[k] = load_A16(ab + k * 32, lda);

    int row0 = 8 * (lane >> 4);
    int coll = lane & 15;
    int ntiles = N >> 4;
    for (int nt = 0; nt < ntiles; ++nt) {
        if (nt + 1 < ntiles)
            __builtin_prefetch(W + (size_t)((nt + 1) * 16) * ldw, 0, 0);
        v8f c = {};
        const _Float16* wb = W + (size_t)(nt * 16) * ldw;
        #pragma unroll
        for (int k = 0; k < KT; ++k) {
            v16h b = load_B16(wb + k * 32, ldw);
            c = wmma_f16(a[k], b, c);
        }
        int colg = nt * 16 + coll;
        float bv = bias[colg];
        #pragma unroll
        for (int r = 0; r < 8; ++r) {
            int rowg = mbase + row0 + r;
            float v = c[r] + bv;
            if (resid)  v += resid[(size_t)rowg * N + colg];
            if (relu)   v = fmaxf(v, 0.0f);
            if (outF32) outF32[(size_t)rowg * N + colg] = v;
            if (outF16) outF16[(size_t)rowg * N + colg] = (_Float16)v;
            if (vt && colg >= 256) {                 // V part of QKV, store transposed
                int cv = colg - 256;
                int hh = cv >> 5, dd = cv & 31;
                int bb = rowg >> 9, nn = rowg & 511;
                vt[(((size_t)bb * NH + hh) * DHD + dd) * NPG + nn] = (_Float16)v;
            }
        }
    }
}

// ---- attention: one wave = one (b, h, 16-query tile); full 512-key softmax via LDS ----
#define SLD 520   // padded LDS row stride (elements)
__global__ void k_attn(const _Float16* __restrict__ qkvh,  // [NT,384] f16
                       const _Float16* __restrict__ vt,    // [B,H,DH,N] f16
                       _Float16* __restrict__ oh) {        // [NT,128] f16
    __shared__ __align__(16) float    sS[16 * SLD];
    __shared__ __align__(16) _Float16 sP[16 * SLD];
#ifdef ATTN_ASYNC_V
    __shared__ __align__(16) _Float16 sV[32 * SLD];        // V slice staged via async DMA
#endif
    int qt = blockIdx.x & 31;
    int h  = (blockIdx.x >> 5) & 3;
    int b  = blockIdx.x >> 7;
    int lane = threadIdx.x;

    const float scale = 0.17677669529663687f;  // 1/sqrt(32)
    const _Float16* qb = qkvh + (size_t)(b * NPG + qt * 16) * 384 + h * DHD;
    const _Float16* kb = qkvh + (size_t)(b * NPG) * 384 + CH + h * DHD;
    const _Float16* vb = vt + ((size_t)(b * NH + h) * DHD) * NPG;   // [32][512]

#ifdef ATTN_ASYNC_V
    // Kick off async global->LDS DMA of the whole V slice (32 x 512 f16) so it
    // overlaps the score WMMAs and the softmax pass. Tracked by ASYNCcnt.
    {
        const _Float16* g0 = vb + lane * 8;          // 16B per lane per issue
        _Float16*       l0 = sV + lane * 8;
        #pragma unroll 4
        for (int r = 0; r < 32; ++r) {
            __builtin_amdgcn_global_load_async_to_lds_b128(
                GPTR(g0 + (size_t)r * NPG), LPTR(l0 + r * SLD), 0, 0);
            __builtin_amdgcn_global_load_async_to_lds_b128(
                GPTR(g0 + (size_t)r * NPG + 256), LPTR(l0 + r * SLD + 256), 0, 0);
        }
    }
#endif

    v16h aq = load_A16(qb, 384);

    int half = lane >> 4;
    int row0 = 8 * half;
    int coll = lane & 15;

    // scores -> LDS (f32)
    for (int kt = 0; kt < 16; ++kt) {
        const _Float16* kt0 = kb + (size_t)(kt * 32) * 384;
        v16h b0 = load_B16(kt0, 384);
        v16h b1 = load_B16(kt0 + (size_t)16 * 384, 384);
        v8f c0 = {}, c1 = {};
        c0 = wmma_f16(aq, b0, c0);
        c1 = wmma_f16(aq, b1, c1);
        int col = kt * 32 + coll;
        #pragma unroll
        for (int r = 0; r < 8; ++r) {
            sS[(row0 + r) * SLD + col]      = c0[r] * scale;
            sS[(row0 + r) * SLD + col + 16] = c1[r] * scale;
        }
    }
    __syncthreads();

    // softmax: each lane owns (row = lane&15, half = lane>>4) -> 256 columns
    int srow = lane & 15;
    const float* srp = sS + srow * SLD + half * 256;
    float m = -1e30f;
    for (int j = 0; j < 256; ++j) m = fmaxf(m, srp[j]);
    m = fmaxf(m, __shfl_xor(m, 16));
    float s = 0.0f;
    _Float16* spp = sP + srow * SLD + half * 256;
    for (int j = 0; j < 256; ++j) {
        float p = __expf(srp[j] - m);
        s += p;
        spp[j] = (_Float16)p;
    }
    s += __shfl_xor(s, 16);
    float inv = 1.0f / s;
    __syncthreads();

#ifdef ATTN_ASYNC_V
    // DMA must have landed before we consume sV (single wave: wait suffices).
#if __has_builtin(__builtin_amdgcn_s_wait_asynccnt)
    __builtin_amdgcn_s_wait_asynccnt(0);
#else
    asm volatile("s_wait_asynccnt 0x0" ::: "memory");
#endif
    const _Float16* vsrc = sV;
    const int vld = SLD;
#else
    const _Float16* vsrc = vb;
    const int vld = NPG;
#endif

    // O = P @ V   (A from LDS, B = rows of [dh][keys] V slice)
    v8f o0 = {}, o1 = {};
    for (int kt = 0; kt < 16; ++kt) {
        v16h ap  = load_A16(sP + kt * 32, SLD);
        v16h bv0 = load_B16(vsrc + kt * 32, vld);
        v16h bv1 = load_B16(vsrc + (size_t)16 * vld + kt * 32, vld);
        o0 = wmma_f16(ap, bv0, o0);
        o1 = wmma_f16(ap, bv1, o1);
    }

    #pragma unroll
    for (int r = 0; r < 8; ++r) {
        float iv = __shfl(inv, row0 + r);   // lane (row0+r) holds 1/sum for that row
        int node = b * NPG + qt * 16 + row0 + r;
        int colb = h * DHD + coll;
        oh[(size_t)node * CH + colb]      = (_Float16)(o0[r] * iv);
        oh[(size_t)node * CH + colb + 16] = (_Float16)(o1[r] * iv);
    }
}

// ---- column sum / sumsq for BN batch statistics (128 columns) ----
__global__ void k_col_stats(const float* __restrict__ Y, float* __restrict__ st) {
    int col = threadIdx.x;                    // 128 threads
    size_t r0 = (size_t)blockIdx.x * 128;     // 256 blocks * 128 rows
    float s = 0.0f, q = 0.0f;
    for (int r = 0; r < 128; ++r) {
        float v = Y[(r0 + r) * CH + col];
        s += v; q += v * v;
    }
    atomicAdd(&st[col], s);
    atomicAdd(&st[CH + col], q);
}

__device__ __forceinline__ float bn_apply(float v, const float* st,
                                          const float* g, const float* bb, int col) {
    float mean = st[col] * (1.0f / NTOT);
    float var  = st[CH + col] * (1.0f / NTOT) - mean * mean;
    return (v - mean) * rsqrtf(var + EPSBN) * g[col] + bb[col];
}

// ---- Z = bn1(Yl) + bn2(Ya); also f16 copy for MLP GEMM ----
__global__ void k_combine(const float* __restrict__ Yl, const float* __restrict__ Ya,
                          const float* __restrict__ st1, const float* __restrict__ st2,
                          const float* __restrict__ g1, const float* __restrict__ b1,
                          const float* __restrict__ g2, const float* __restrict__ b2,
                          float* __restrict__ Z, _Float16* __restrict__ Zh) {
    int i = blockIdx.x * blockDim.x + threadIdx.x;
    if (i >= NTOT * CH) return;
    int col = i & (CH - 1);
    float z = bn_apply(Yl[i], st1, g1, b1, col) + bn_apply(Ya[i], st2, g2, b2, col);
    Z[i] = z;
    Zh[i] = (_Float16)z;
}

// ---- out = bn3(F) ----
__global__ void k_final(const float* __restrict__ F, const float* __restrict__ st3,
                        const float* __restrict__ g3, const float* __restrict__ b3,
                        float* __restrict__ out) {
    int i = blockIdx.x * blockDim.x + threadIdx.x;
    if (i >= NTOT * CH) return;
    int col = i & (CH - 1);
    out[i] = bn_apply(F[i], st3, g3, b3, col);
}

extern "C" void kernel_launch(void* const* d_in, const int* in_sizes, int n_in,
                              void* d_out, int out_size, void* d_ws, size_t ws_size,
                              hipStream_t stream) {
    const float* x   = (const float*)d_in[0];
    const int*   ei  = (const int*)  d_in[1];
    const float* Wc  = (const float*)d_in[2];
    const float* bc  = (const float*)d_in[3];
    const float* ipw = (const float*)d_in[4];
    const float* ipb = (const float*)d_in[5];
    const float* opw = (const float*)d_in[6];
    const float* opb = (const float*)d_in[7];
    const float* g1  = (const float*)d_in[8];
    const float* b1  = (const float*)d_in[9];
    const float* g2  = (const float*)d_in[10];
    const float* b2  = (const float*)d_in[11];
    const float* g3  = (const float*)d_in[12];
    const float* b3  = (const float*)d_in[13];
    const float* Wm1 = (const float*)d_in[14];
    const float* bm1 = (const float*)d_in[15];
    const float* Wm2 = (const float*)d_in[16];
    const float* bm2 = (const float*)d_in[17];
    float* out = (float*)d_out;

    // ---- workspace layout ----
    char* p = (char*)d_ws;
    auto take = [&](size_t bytes) { char* r = p; p += (bytes + 255) & ~(size_t)255; return r; };
    float*    agg  = (float*)   take((size_t)NTOT * CH * 4);
    float*    cnt  = (float*)   take((size_t)NTOT * 4);
    float*    st   = (float*)   take(768 * 4);           // st1 | st2 | st3
    _Float16* xh   = (_Float16*)take((size_t)NTOT * CH * 2);
    _Float16* ah   = (_Float16*)take((size_t)NTOT * CH * 2);
    _Float16* wch  = (_Float16*)take((size_t)CH * CH * 2);
    _Float16* iph  = (_Float16*)take((size_t)3 * CH * CH * 2);
    _Float16* oph  = (_Float16*)take((size_t)CH * CH * 2);
    _Float16* wm1h = (_Float16*)take((size_t)2 * CH * CH * 2);
    _Float16* wm2h = (_Float16*)take((size_t)2 * CH * CH * 2);
    _Float16* qkvh = (_Float16*)take((size_t)NTOT * 3 * CH * 2);
    _Float16* vth  = (_Float16*)take((size_t)NTOT * CH * 2);
    _Float16* oh   = (_Float16*)take((size_t)NTOT * CH * 2);
    float*    Yl   = (float*)   take((size_t)NTOT * CH * 4);
    float*    Ya   = (float*)   take((size_t)NTOT * CH * 4);
    float*    Z    = (float*)   take((size_t)NTOT * CH * 4);
    // reuse: Zh aliases vth (dead after attention), Uh aliases qkvh, F aliases Yl
    _Float16* Zh = vth;
    _Float16* Uh = qkvh;
    float*    F  = Yl;
    float* st1 = st, *st2 = st + 256, *st3 = st + 512;

    // zero accumulators (agg + cnt + stats are contiguous)
    (void)hipMemsetAsync(agg, 0, (size_t)NTOT * CH * 4 + (size_t)NTOT * 4 + 768 * 4, stream);

    // convert activations / weights to f16
    int n;
    n = NTOT * CH;      k_cvt_f16<<<(n + 255) / 256, 256, 0, stream>>>(x,   xh,   n);
    n = CH * CH;        k_cvt_f16<<<(n + 255) / 256, 256, 0, stream>>>(Wc,  wch,  n);
    n = 3 * CH * CH;    k_cvt_f16<<<(n + 255) / 256, 256, 0, stream>>>(ipw, iph,  n);
    n = CH * CH;        k_cvt_f16<<<(n + 255) / 256, 256, 0, stream>>>(opw, oph,  n);
    n = 2 * CH * CH;    k_cvt_f16<<<(n + 255) / 256, 256, 0, stream>>>(Wm1, wm1h, n);
    n = 2 * CH * CH;    k_cvt_f16<<<(n + 255) / 256, 256, 0, stream>>>(Wm2, wm2h, n);

    // local branch: scatter-mean
    k_scatter<<<(NEDGE * 4) / 256, 256, 0, stream>>>(ei, x, agg, cnt);
    k_agg_fin<<<(NTOT * CH) / 256, 256, 0, stream>>>(agg, cnt, ah);

    const int GB = (NTOT / 16) / 4;   // 512 blocks of 4 waves
    // Yl = ah @ Wc^T + bc + x
    k_gemm<4><<<GB, 128, 0, stream>>>(ah, CH, wch, CH, bc, x, Yl, nullptr, nullptr, CH, 0);
    // qkvh = xh @ in_proj^T + b  (V also stored transposed to vth)
    k_gemm<4><<<GB, 128, 0, stream>>>(xh, CH, iph, CH, ipb, nullptr, nullptr, qkvh, vth, 3 * CH, 0);
    // attention
    k_attn<<<BGR * NH * (NPG / 16), 32, 0, stream>>>(qkvh, vth, oh);
    // Ya = oh @ out_proj^T + b + x
    k_gemm<4><<<GB, 128, 0, stream>>>(oh, CH, oph, CH, opb, x, Ya, nullptr, nullptr, CH, 0);

    // BN stats for both branches
    k_col_stats<<<256, 128, 0, stream>>>(Yl, st1);
    k_col_stats<<<256, 128, 0, stream>>>(Ya, st2);

    // Z = bn1(Yl) + bn2(Ya)
    k_combine<<<(NTOT * CH) / 256, 256, 0, stream>>>(Yl, Ya, st1, st2, g1, b1, g2, b2, Z, Zh);

    // MLP: Uh = relu(Zh @ Wm1^T + bm1) ; F = Uh @ Wm2^T + bm2 + Z
    k_gemm<4><<<GB, 128, 0, stream>>>(Zh, CH, wm1h, CH, bm1, nullptr, nullptr, Uh, nullptr, 2 * CH, 1);
    k_gemm<8><<<GB, 128, 0, stream>>>(Uh, 2 * CH, wm2h, 2 * CH, bm2, Z, F, nullptr, nullptr, CH, 0);

    // final BN
    k_col_stats<<<256, 128, 0, stream>>>(F, st3);
    k_final<<<(NTOT * CH) / 256, 256, 0, stream>>>(F, st3, g3, b3, out);
}
